// Self_Attention_12214886990752
// MI455X (gfx1250) — matrix operand order repaired and checked by reading, not compiled, not run on hardware
//
#include <hip/hip_runtime.h>
#include <hip/hip_bf16.h>

// Problem constants (match reference)
#define BATCH 8
#define SEQ   2048
#define DIM   512
#define KD    64
#define BS    (BATCH * SEQ)   // 16384
#define QSPLIT 4              // stats q-axis split factor

typedef __bf16 bf16_t;
typedef __attribute__((ext_vector_type(16))) __bf16 v16bf;
typedef __attribute__((ext_vector_type(8)))  __bf16 bf16x8;
typedef __attribute__((ext_vector_type(8)))  float  v8f;

#define V8F_ZERO ((v8f){0.f,0.f,0.f,0.f,0.f,0.f,0.f,0.f})

// ---------------------------------------------------------------------------
// WMMA wrapper: D = A(16x32 bf16) x B(32x16 bf16) + C(16x16 f32)
// ---------------------------------------------------------------------------
static __device__ __forceinline__ v8f wmma_bf16(v16bf a, v16bf b, v8f c) {
  return __builtin_amdgcn_wmma_f32_16x16x32_bf16(
      /*neg_a=*/false, a, /*neg_b=*/false, b,
      /*c_mod=*/(short)0, c, /*reuse_a=*/false, /*reuse_b=*/false);
}

// ---------------------------------------------------------------------------
// Fragment loaders (wave32 layouts per CDNA5 ISA 7.12.2)
// A 16x32 bf16: lane m = lane&15; K elems kb..kb+7, kb+16..kb+23, kb=(lane&16)?8:0
// ---------------------------------------------------------------------------
static __device__ __forceinline__ v16bf frag_a_bf16(const bf16_t* tile, int ld) {
  int lane = threadIdx.x & 31;
  int m    = lane & 15;
  int kb   = (lane & 16) ? 8 : 0;
  const bf16_t* p = tile + (long)m * ld + kb;
  bf16x8 lo = *(const bf16x8*)(p);
  bf16x8 hi = *(const bf16x8*)(p + 16);
  v16bf a;
#pragma unroll
  for (int i = 0; i < 8; ++i) { a[i] = lo[i]; a[8 + i] = hi[i]; }
  return a;
}

// A fragment straight from fp32 row-major source (cast to bf16 in regs)
static __device__ __forceinline__ v16bf frag_a_f32(const float* tile, int ld) {
  int lane = threadIdx.x & 31;
  int m    = lane & 15;
  int kb   = (lane & 16) ? 8 : 0;
  const float* p = tile + (long)m * ld + kb;
  float4 l0 = *reinterpret_cast<const float4*>(p);
  float4 l1 = *reinterpret_cast<const float4*>(p + 4);
  float4 h0 = *reinterpret_cast<const float4*>(p + 16);
  float4 h1 = *reinterpret_cast<const float4*>(p + 20);
  v16bf a;
  a[0]  = (bf16_t)l0.x; a[1]  = (bf16_t)l0.y; a[2]  = (bf16_t)l0.z; a[3]  = (bf16_t)l0.w;
  a[4]  = (bf16_t)l1.x; a[5]  = (bf16_t)l1.y; a[6]  = (bf16_t)l1.z; a[7]  = (bf16_t)l1.w;
  a[8]  = (bf16_t)h0.x; a[9]  = (bf16_t)h0.y; a[10] = (bf16_t)h0.z; a[11] = (bf16_t)h0.w;
  a[12] = (bf16_t)h1.x; a[13] = (bf16_t)h1.y; a[14] = (bf16_t)h1.z; a[15] = (bf16_t)h1.w;
  return a;
}

// B 32x16 bf16: lane n = lane&15; K elems kb..kb+7, kb+16..kb+23, kb=(lane&16)?8:0.
// Source is N-major [N][ld] (transposed, so K contiguous per lane).
static __device__ __forceinline__ v16bf frag_b_nmajor(const bf16_t* tile, int ld) {
  int lane = threadIdx.x & 31;
  int n    = lane & 15;
  int kb   = (lane & 16) ? 8 : 0;
  const bf16_t* p = tile + (long)n * ld + kb;
  bf16x8 lo = *(const bf16x8*)(p);
  bf16x8 hi = *(const bf16x8*)(p + 16);
  v16bf b;
#pragma unroll
  for (int i = 0; i < 8; ++i) { b[i] = lo[i]; b[8 + i] = hi[i]; }
  return b;
}

// Per-lane prefetch of the cacheline(s) a B-fragment will read (one line/lane,
// lanes cover all 16 rows of the fragment tile).
static __device__ __forceinline__ void prefetch_b_nmajor(const bf16_t* tile, int ld) {
  int lane = threadIdx.x & 31;
  int n    = lane & 15;
  int kb   = (lane & 16) ? 16 : 0;
  __builtin_prefetch(tile + (long)n * ld + kb, 0, 1);
}

// ---------------------------------------------------------------------------
// Kernel 0a: weights [512,64] fp32 -> N-major bf16 [64,512]
// ---------------------------------------------------------------------------
__global__ void prep_weights(const float* __restrict__ Wk, const float* __restrict__ Wq,
                             bf16_t* __restrict__ WkT, bf16_t* __restrict__ WqT) {
  for (int idx = threadIdx.x; idx < DIM * KD; idx += blockDim.x) {
    int d = idx >> 6, n = idx & 63;
    WkT[n * DIM + d] = (bf16_t)Wk[idx];
    WqT[n * DIM + d] = (bf16_t)Wq[idx];
  }
}

// ---------------------------------------------------------------------------
// Kernel 0b: VT[b,d,s] = bf16(X[b,s,d])  (LDS-tiled 32x32 transpose)
// ---------------------------------------------------------------------------
__global__ void prep_vt(const float* __restrict__ X, bf16_t* __restrict__ VT) {
  __shared__ float tile[32][33];
  int bt = blockIdx.x;
  int b  = bt >> 10;                 // (SEQ/32)*(DIM/32) = 1024 tiles per batch
  int t  = bt & 1023;
  int s0 = (t >> 4) << 5;
  int d0 = (t & 15) << 5;
  for (int i = threadIdx.x; i < 1024; i += blockDim.x) {
    int r = i >> 5, c = i & 31;
    tile[r][c] = X[((long)b * SEQ + s0 + r) * DIM + d0 + c];
  }
  __syncthreads();
  for (int i = threadIdx.x; i < 1024; i += blockDim.x) {
    int r = i >> 5, c = i & 31;
    VT[((long)b * DIM + d0 + r) * SEQ + s0 + c] = (bf16_t)tile[c][r];
  }
}

// ---------------------------------------------------------------------------
// Kernel 1: Q/K projection.  [BS,512] @ [512,64] (+bias) -> bf16 [BS,64] x2
// ---------------------------------------------------------------------------
__global__ void proj_kernel(const float* __restrict__ X,
                            const bf16_t* __restrict__ WqT, const bf16_t* __restrict__ WkT,
                            const float* __restrict__ bq,  const float* __restrict__ bk,
                            bf16_t* __restrict__ Qh, bf16_t* __restrict__ Kh) {
  int wave = threadIdx.x >> 5;
  int lane = threadIdx.x & 31;
  long r0  = (long)blockIdx.x * 64 + wave * 16;

  v8f accQ[4], accK[4];
#pragma unroll
  for (int nt = 0; nt < 4; ++nt) { accQ[nt] = V8F_ZERO; accK[nt] = V8F_ZERO; }

  for (int kc = 0; kc < DIM; kc += 32) {
    // prefetch next k-chunk of X rows (per-lane, covers upcoming lines)
    if (kc + 32 < DIM) {
      int m  = lane & 15;
      int kb = (lane & 16) ? 8 : 0;
      __builtin_prefetch(X + r0 * DIM + (long)m * DIM + kc + 32 + kb, 0, 1);
    }
    // ---- load phase: A + all 8 weight B-fragments ----
    v16bf a = frag_a_f32(X + r0 * DIM + kc, DIM);
    v16bf wq[4], wk[4];
#pragma unroll
    for (int nt = 0; nt < 4; ++nt) {
      wq[nt] = frag_b_nmajor(WqT + (nt * 16) * DIM + kc, DIM);
      wk[nt] = frag_b_nmajor(WkT + (nt * 16) * DIM + kc, DIM);
    }
    // ---- compute phase: 8 back-to-back WMMAs ----
#pragma unroll
    for (int nt = 0; nt < 4; ++nt) {
      accQ[nt] = wmma_bf16(a, wq[nt], accQ[nt]);
      accK[nt] = wmma_bf16(a, wk[nt], accK[nt]);
    }
  }

  int n16 = lane & 15, mb = (lane & 16) ? 8 : 0;
#pragma unroll
  for (int nt = 0; nt < 4; ++nt) {
    int n = nt * 16 + n16;
    float biasq = bq[n], biask = bk[n];
#pragma unroll
    for (int i = 0; i < 8; ++i) {
      long row = r0 + mb + i;
      Qh[row * KD + n] = (bf16_t)(accQ[nt][i] + biasq);
      Kh[row * KD + n] = (bf16_t)(accK[nt][i] + biask);
    }
  }
}

// ---------------------------------------------------------------------------
// Kernel 2: per-s-column softmax stats over the q axis (online max/sum),
// q-range split QSPLIT ways (grid.y); partials combined after.
// ---------------------------------------------------------------------------
__global__ void stats_kernel(const bf16_t* __restrict__ Qh, const bf16_t* __restrict__ Kh,
                             float* __restrict__ mPart, float* __restrict__ zPart) {
  int wave = threadIdx.x >> 5;
  int lane = threadIdx.x & 31;
  long s0  = (long)blockIdx.x * 128 + wave * 16;   // flat b*SEQ + s
  long b   = s0 >> 11;
  int  qp  = blockIdx.y;

  v16bf bk0 = frag_b_nmajor(Kh + s0 * KD + 0,  KD);
  v16bf bk1 = frag_b_nmajor(Kh + s0 * KD + 32, KD);

  const int TILES = (SEQ / QSPLIT) / 16;           // 32
  int tbeg = qp * TILES;
  long qrow0 = b * SEQ + (long)tbeg * 16;

  float m = -__builtin_inff();
  float Z = 0.f;

  v16bf a0 = frag_a_bf16(Qh + qrow0 * KD + 0,  KD);
  v16bf a1 = frag_a_bf16(Qh + qrow0 * KD + 32, KD);

  for (int t = 0; t < TILES; ++t) {
    // prefetch next tile's A fragments (overlaps with WMMA + softmax VALU)
    int tn = (t + 1 < TILES) ? (t + 1) : t;
    long qrn = b * SEQ + (long)(tbeg + tn) * 16;
    v16bf na0 = frag_a_bf16(Qh + qrn * KD + 0,  KD);
    v16bf na1 = frag_a_bf16(Qh + qrn * KD + 32, KD);

    v8f cf = V8F_ZERO;
    cf = wmma_bf16(a0, bk0, cf);
    cf = wmma_bf16(a1, bk1, cf);

    float tm = cf[0];
#pragma unroll
    for (int i = 1; i < 8; ++i) tm = fmaxf(tm, cf[i]);
    tm = fmaxf(tm, __shfl_xor(tm, 16, 32));
    float mn = fmaxf(m, tm);
    float part = 0.f;
#pragma unroll
    for (int i = 0; i < 8; ++i) part += __expf(cf[i] - mn);
    part += __shfl_xor(part, 16, 32);
    Z = Z * __expf(m - mn) + part;
    m = mn;

    a0 = na0; a1 = na1;
  }
  if (lane < 16) {
    mPart[(long)qp * BS + s0 + lane] = m;
    zPart[(long)qp * BS + s0 + lane] = Z;
  }
}

__global__ void stats_combine(const float* __restrict__ mPart, const float* __restrict__ zPart,
                              float* __restrict__ colMax, float* __restrict__ colSum) {
  int i = blockIdx.x * 256 + threadIdx.x;
  if (i >= BS) return;
  float m = mPart[i];
#pragma unroll
  for (int p = 1; p < QSPLIT; ++p) m = fmaxf(m, mPart[(long)p * BS + i]);
  float Z = 0.f;
#pragma unroll
  for (int p = 0; p < QSPLIT; ++p)
    Z += zPart[(long)p * BS + i] * __expf(mPart[(long)p * BS + i] - m);
  colMax[i] = m;
  colSum[i] = Z;
}

// ---------------------------------------------------------------------------
// Kernel 3: fused  out[q,:] = sum_s exp(l[q,s]-m[s])/Z[s] * V[s,:]
// 16 waves, q-block 64, s-chunk 64.  Per iteration:
//   1. prefetch this chunk's VT lines (overlaps producer + barrier wait)
//   2. every wave produces one 16x16 logit tile -> exp -> double-buffered LDS
//   3. one barrier
//   4. consume 16 WMMA vs VT (fragments reused across the wave's two q-tiles);
//      prefetch next chunk's Kh producer lines during consume.
// ---------------------------------------------------------------------------
__global__ void __launch_bounds__(512)
out_kernel(const bf16_t* __restrict__ Qh, const bf16_t* __restrict__ Kh,
           const bf16_t* __restrict__ VT,
           const float* __restrict__ colMax, const float* __restrict__ colSum,
           float* __restrict__ out) {
  __shared__ bf16_t Pbuf[2][64 * 64];

  int wave = threadIdx.x >> 5;
  int lane = threadIdx.x & 31;
  int b    = blockIdx.x >> 5;                  // SEQ/64 = 32 q-blocks per batch
  int q0   = (blockIdx.x & 31) * 64;
  long qbase = (long)b * SEQ + q0;

  // producer tile: qt in 0..3 (q sub-tile), st in 0..3 (s sub-tile)
  int qt = wave & 3;
  int st = wave >> 2;
  v16bf aq0 = frag_a_bf16(Qh + (qbase + qt * 16) * KD + 0,  KD);
  v16bf aq1 = frag_a_bf16(Qh + (qbase + qt * 16) * KD + 32, KD);

  // consumer assignment: d segment + q half
  int d0 = (wave & 7) * 64;
  int qh = wave >> 3;

  v8f acc[2][4];
#pragma unroll
  for (int q = 0; q < 2; ++q)
#pragma unroll
    for (int dt = 0; dt < 4; ++dt) acc[q][dt] = V8F_ZERO;

  int n16 = lane & 15, mb = (lane & 16) ? 8 : 0;

  for (int it = 0; it < SEQ / 64; ++it) {
    int sc = it * 64;
    bf16_t* Pb = Pbuf[it & 1];

    // ---- 1. warm this chunk's VT lines before the barrier ---------------
#pragma unroll
    for (int dt = 0; dt < 4; ++dt)
      prefetch_b_nmajor(VT + ((long)b * DIM + d0 + dt * 16) * SEQ + sc, SEQ);

    // ---- 2. produce: 16x16 logit tile -> exp -> LDS ---------------------
    long srow = (long)b * SEQ + sc + st * 16;
    v16bf kb0 = frag_b_nmajor(Kh + srow * KD + 0,  KD);
    v16bf kb1 = frag_b_nmajor(Kh + srow * KD + 32, KD);
    v8f cf = V8F_ZERO;
    cf = wmma_bf16(aq0, kb0, cf);
    cf = wmma_bf16(aq1, kb1, cf);

    int sl = st * 16 + n16;
    long sidx = (long)b * SEQ + sc + sl;
    float mC = colMax[sidx];
    float rz = 1.0f / colSum[sidx];
#pragma unroll
    for (int i = 0; i < 8; ++i) {
      float p = __expf(cf[i] - mC) * rz;
      Pb[(qt * 16 + mb + i) * 64 + sl] = (bf16_t)p;
    }
    // ---- 3. single barrier (double buffer handles WAR across iters) -----
    __syncthreads();

    // ---- 4. consume: P(32x64) x VT(64x64) -------------------------------
    // prefetch next chunk's producer (Kh) lines; overlaps the WMMA chain
    if (it + 1 < SEQ / 64)
      prefetch_b_nmajor(Kh + ((long)b * SEQ + sc + 64 + st * 16) * KD, KD);

    const bf16_t* Pq = Pb + (qh * 32) * 64;
    v16bf ap0 = frag_a_bf16(Pq, 64);
    v16bf ap1 = frag_a_bf16(Pq + 32, 64);
    v16bf ap2 = frag_a_bf16(Pq + 16 * 64, 64);
    v16bf ap3 = frag_a_bf16(Pq + 16 * 64 + 32, 64);
#pragma unroll
    for (int dt = 0; dt < 4; ++dt) {
      const bf16_t* vt = VT + ((long)b * DIM + d0 + dt * 16) * SEQ + sc;
      v16bf bv0 = frag_b_nmajor(vt, SEQ);
      v16bf bv1 = frag_b_nmajor(vt + 32, SEQ);
      acc[0][dt] = wmma_bf16(ap0, bv0, acc[0][dt]);
      acc[0][dt] = wmma_bf16(ap1, bv1, acc[0][dt]);
      acc[1][dt] = wmma_bf16(ap2, bv0, acc[1][dt]);
      acc[1][dt] = wmma_bf16(ap3, bv1, acc[1][dt]);
    }
  }

  // ---------------- epilogue ----------------
#pragma unroll
  for (int q = 0; q < 2; ++q)
#pragma unroll
    for (int dt = 0; dt < 4; ++dt)
#pragma unroll
      for (int i = 0; i < 8; ++i) {
        long qq = qbase + qh * 32 + q * 16 + mb + i;
        out[qq * DIM + d0 + dt * 16 + n16] = acc[q][dt][i];
      }
}

// ---------------------------------------------------------------------------
// Launch: prep -> proj -> stats(+combine) -> fused output. All stream-ordered.
// Workspace (~20.9 MB): Qh | Kh | VT | WqT | WkT | mPart | zPart | colMax | colSum
// ---------------------------------------------------------------------------
extern "C" void kernel_launch(void* const* d_in, const int* in_sizes, int n_in,
                              void* d_out, int out_size, void* d_ws, size_t ws_size,
                              hipStream_t stream) {
  (void)in_sizes; (void)n_in; (void)out_size; (void)ws_size;

  const float* X  = (const float*)d_in[0];   // inputs       [B,S,D]
  const float* Wk = (const float*)d_in[1];   // key_extract  [D,K]
  const float* bk = (const float*)d_in[2];   // key_bias     [1,K]
  const float* Wq = (const float*)d_in[3];   // query_extract[D,K]
  const float* bq = (const float*)d_in[4];   // query_bias   [1,K]
  float* out = (float*)d_out;

  char* ws = (char*)d_ws;
  size_t off = 0;
  bf16_t* Qh  = (bf16_t*)(ws + off); off += (size_t)BS * KD * sizeof(bf16_t);
  bf16_t* Kh  = (bf16_t*)(ws + off); off += (size_t)BS * KD * sizeof(bf16_t);
  bf16_t* VT  = (bf16_t*)(ws + off); off += (size_t)BATCH * DIM * SEQ * sizeof(bf16_t);
  bf16_t* WqT = (bf16_t*)(ws + off); off += (size_t)KD * DIM * sizeof(bf16_t);
  bf16_t* WkT = (bf16_t*)(ws + off); off += (size_t)KD * DIM * sizeof(bf16_t);
  float* mPart  = (float*)(ws + off); off += (size_t)QSPLIT * BS * sizeof(float);
  float* zPart  = (float*)(ws + off); off += (size_t)QSPLIT * BS * sizeof(float);
  float* colMax = (float*)(ws + off); off += (size_t)BS * sizeof(float);
  float* colSum = (float*)(ws + off); off += (size_t)BS * sizeof(float);

  prep_weights<<<1, 256, 0, stream>>>(Wk, Wq, WkT, WqT);
  prep_vt<<<BATCH * (SEQ / 32) * (DIM / 32), 256, 0, stream>>>(X, VT);
  proj_kernel<<<BS / 64, 128, 0, stream>>>(X, WqT, WkT, bq, bk, Qh, Kh);
  stats_kernel<<<dim3(BS / 128, QSPLIT), 256, 0, stream>>>(Qh, Kh, mPart, zPart);
  stats_combine<<<(BS + 255) / 256, 256, 0, stream>>>(mPart, zPart, colMax, colSum);
  out_kernel<<<BATCH * (SEQ / 64), 512, 0, stream>>>(Qh, Kh, VT, colMax, colSum, out);
}
